// PythagorasModel_34617436405985
// MI455X (gfx1250) — compile-verified, matrix operand order checked by reading, not run on hardware
//
#include <hip/hip_runtime.h>
#include <hip/hip_bf16.h>
#include <stdint.h>

// ---------------- types ----------------
typedef __bf16 bf16x16 __attribute__((ext_vector_type(16)));
typedef float  f32x8   __attribute__((ext_vector_type(8)));
typedef unsigned int uint32x4 __attribute__((ext_vector_type(4)));
typedef int int32x8 __attribute__((ext_vector_type(8)));
typedef int int32x4 __attribute__((ext_vector_type(4)));

__device__ __forceinline__ unsigned short f2bf(float f) {
  union { float f; unsigned int u; } cv; cv.f = f;
  unsigned int u = cv.u;
  u += 0x7FFFu + ((u >> 16) & 1u);   // round-to-nearest-even
  return (unsigned short)(u >> 16);
}

#define BM 128
#define BN 128
#define BK 32
#define LDSS 40   // LDS row stride in ushorts (80 B = 64 B data + 16 B pad)

// ---- TDM: load a 128-row x 32-col bf16 tile (row-major, row stride Ka elems)
// from global into LDS with 16B padding after every 64B row -> 80B LDS rows.
__device__ __forceinline__ void tdm_load_tile(unsigned int lds_byte_addr,
                                              const unsigned short* gbase,
                                              int row0, int k0, int Ka,
                                              int tensor_rows) {
  unsigned long long ga = (unsigned long long)(uintptr_t)gbase +
      ((unsigned long long)(unsigned)row0 * (unsigned)Ka + (unsigned)k0) * 2ull;
  // D# group 0: count=1 | lds_addr | global_addr | type=2
  uint32x4 g0 = { 1u,
                  lds_byte_addr,
                  (unsigned int)(ga & 0xFFFFFFFFull),
                  (unsigned int)(ga >> 32) | (2u << 30) };
  // D# group 1: data_size=2B(code 1), pad_enable, pad_interval=16DW(code 3),
  // pad_amount=4DW(code 3); tensor_dim0=Ka; tensor_dim1=tensor_rows;
  // tile_dim0=32; tile_dim1=128; tensor_dim0_stride=Ka
  unsigned int w0 = (1u << 16) | (1u << 20) | (3u << 22) | (3u << 25);
  unsigned int w1 = ((unsigned int)Ka & 0xFFFFu) << 16;
  unsigned int w2 = ((unsigned int)Ka >> 16) |
                    (((unsigned int)tensor_rows & 0xFFFFu) << 16);
  unsigned int w3 = ((unsigned int)tensor_rows >> 16) | (32u << 16);
  unsigned int w4 = 128u;
  unsigned int w5 = (unsigned int)Ka;
  int32x8 g1 = { (int)w0, (int)w1, (int)w2, (int)w3,
                 (int)w4, (int)w5, 0, 0 };
  int32x4 z4 = { 0, 0, 0, 0 };
  int32x8 z8 = { 0, 0, 0, 0, 0, 0, 0, 0 };
  __builtin_amdgcn_tensor_load_to_lds(g0, g1, z4, z4, z8, 0);
}

// ---------------- WMMA bf16 GEMM (pre-packed bf16 A [M][Ka], Bt [N][Ka]) ----
// Double-buffered TDM pipeline: wave0 DMAs tile kt+1 while all waves compute
// tile kt. All tile loads are unguarded: Ka % 32 == 0, A has >= gridDim.y*128
// rows, Bt has >= gridDim.x*128 rows. Store guards handle real M/N.
__global__ __launch_bounds__(256) void gemm_bf16_wmma(
    const unsigned short* __restrict__ A, const unsigned short* __restrict__ Bt,
    const float* __restrict__ bias, void* __restrict__ Cout,
    int Mstore, int Nstore, int Ka, int Arows, int Brows,
    int do_relu, int has_bias, int out_bf16)
{
  __shared__ alignas(16) unsigned short sA[2][BM * LDSS];
  __shared__ alignas(16) unsigned short sB[2][BN * LDSS];

  const int tid   = threadIdx.x;
  const int lane  = tid & 31;
  const int wave  = tid >> 5;
  const int waveM = wave >> 2;   // 0..1 -> 64 rows each
  const int waveN = wave & 3;    // 0..3 -> 32 cols each
  const int r  = lane & 15;
  const int hi = lane >> 4;

  const int blockN0 = blockIdx.x * BN;
  const int blockM0 = blockIdx.y * BM;

  unsigned int aAddr[2], bAddr[2];
  aAddr[0] = (unsigned int)(uintptr_t)(&sA[0][0]);
  aAddr[1] = (unsigned int)(uintptr_t)(&sA[1][0]);
  bAddr[0] = (unsigned int)(uintptr_t)(&sB[0][0]);
  bAddr[1] = (unsigned int)(uintptr_t)(&sB[1][0]);

  f32x8 acc[4][2];
  for (int i = 0; i < 4; ++i)
    for (int j = 0; j < 2; ++j)
      acc[i][j] = {};

  const int ktiles = Ka / BK;

  // prefill buffer 0 with tile 0
  if (wave == 0) {
    tdm_load_tile(aAddr[0], A,  blockM0, 0, Ka, Arows);
    tdm_load_tile(bAddr[0], Bt, blockN0, 0, Ka, Brows);
    __builtin_amdgcn_s_wait_tensorcnt(0);
  }
  __syncthreads();

  for (int kt = 0; kt < ktiles; ++kt) {
    const int cur = kt & 1;
    const int nxt = cur ^ 1;

    // prefetch next tile into the alternate buffer (overlaps with compute)
    if (wave == 0 && (kt + 1) < ktiles) {
      const int k1 = (kt + 1) * BK;
      tdm_load_tile(aAddr[nxt], A,  blockM0, k1, Ka, Arows);
      tdm_load_tile(bAddr[nxt], Bt, blockN0, k1, Ka, Brows);
    }

    // fragments per ISA 16-bit A layout:
    // lane<16 holds K=0..7,16..23 ; lane>=16 holds K=8..15,24..31
    union FragU { bf16x16 v; uint4 q[2]; };
    FragU fa[4], fb[2];
    #pragma unroll
    for (int ms = 0; ms < 4; ++ms) {
      const unsigned short* p = &sA[cur][(waveM * 64 + ms * 16 + r) * LDSS + hi * 8];
      fa[ms].q[0] = *(const uint4*)(p);
      fa[ms].q[1] = *(const uint4*)(p + 16);
    }
    #pragma unroll
    for (int ns = 0; ns < 2; ++ns) {
      const unsigned short* p = &sB[cur][(waveN * 32 + ns * 16 + r) * LDSS + hi * 8];
      fb[ns].q[0] = *(const uint4*)(p);
      fb[ns].q[1] = *(const uint4*)(p + 16);
    }
    #pragma unroll
    for (int ms = 0; ms < 4; ++ms)
      #pragma unroll
      for (int ns = 0; ns < 2; ++ns)
        acc[ms][ns] = __builtin_amdgcn_wmma_f32_16x16x32_bf16(
            false, fa[ms].v, false, fb[ns].v, (short)0, acc[ms][ns], false, false);

    // complete the prefetch, then release everyone into the next iteration
    if (wave == 0) __builtin_amdgcn_s_wait_tensorcnt(0);
    __syncthreads();
  }

  // epilogue: C layout VGPR i -> row = i + 8*hi, col = lane&15
  // fully unrolled so acc[][] stays in VGPRs (no scratch demotion)
  if (out_bf16) {
    #pragma unroll
    for (int ms = 0; ms < 4; ++ms) {
      int rowBase = blockM0 + waveM * 64 + ms * 16 + hi * 8;
      #pragma unroll
      for (int ns = 0; ns < 2; ++ns) {
        int col = blockN0 + waveN * 32 + ns * 16 + r;
        if (col >= Nstore) continue;
        float bv = has_bias ? bias[col] : 0.0f;
        #pragma unroll
        for (int i = 0; i < 8; ++i) {
          int row = rowBase + i;
          if (row >= Mstore) continue;
          float v = acc[ms][ns][i] + bv;
          if (do_relu) v = fmaxf(v, 0.0f);
          ((unsigned short*)Cout)[(size_t)row * Nstore + col] = f2bf(v);
        }
      }
    }
  } else {
    #pragma unroll
    for (int ms = 0; ms < 4; ++ms) {
      int rowBase = blockM0 + waveM * 64 + ms * 16 + hi * 8;
      #pragma unroll
      for (int ns = 0; ns < 2; ++ns) {
        int col = blockN0 + waveN * 32 + ns * 16 + r;
        if (col >= Nstore) continue;
        float bv = has_bias ? bias[col] : 0.0f;
        #pragma unroll
        for (int i = 0; i < 8; ++i) {
          int row = rowBase + i;
          if (row >= Mstore) continue;
          float v = acc[ms][ns][i] + bv;
          if (do_relu) v = fmaxf(v, 0.0f);
          ((float*)Cout)[(size_t)row * Nstore + col] = v;
        }
      }
    }
  }
}

// ---------------- pack / transpose kernels (f32 -> bf16, zero padded) -------
__global__ void k_pack_rows(const float* __restrict__ src, unsigned short* dst,
                            int rows, int cols, int prows, int pcols) {
  long long idx = (long long)blockIdx.x * blockDim.x + threadIdx.x;
  long long tot = (long long)prows * pcols;
  if (idx >= tot) return;
  int rr = (int)(idx / pcols);
  int cc = (int)(idx % pcols);
  float v = (rr < rows && cc < cols) ? src[(size_t)rr * cols + cc] : 0.0f;
  dst[idx] = f2bf(v);
}
__global__ void k_pack_transpose(const float* __restrict__ W, unsigned short* Wt,
                                 int K, int N, int pK, int pN) {
  long long idx = (long long)blockIdx.x * blockDim.x + threadIdx.x;
  long long tot = (long long)pN * pK;
  if (idx >= tot) return;
  int nn = (int)(idx / pK);
  int kk = (int)(idx % pK);
  float v = (kk < K && nn < N) ? W[(size_t)kk * N + nn] : 0.0f;
  Wt[idx] = f2bf(v);
}

// ---------------- graph / pooling kernels ----------------
__global__ void k_init_deg(float* deg, int n) {
  int i = blockIdx.x * blockDim.x + threadIdx.x;
  if (i < n) deg[i] = 1.0f;   // self-loop
}
__global__ void k_count_deg(const int* __restrict__ ei, float* deg, int E) {
  int e = blockIdx.x * blockDim.x + threadIdx.x;
  if (e < E) atomicAdd(&deg[ei[E + e]], 1.0f);  // col = target
}
__global__ void k_rsqrt(const float* __restrict__ deg, float* dinv, int n) {
  int i = blockIdx.x * blockDim.x + threadIdx.x;
  if (i < n) { float d = deg[i]; dinv[i] = d > 0.f ? rsqrtf(d) : 0.f; }
}
__global__ void k_self_bias(const float* __restrict__ hw, const float* __restrict__ dinv,
                            const float* __restrict__ bias, float* __restrict__ out,
                            int n, int H) {
  long long idx = (long long)blockIdx.x * blockDim.x + threadIdx.x;
  long long tot = (long long)n * H;
  if (idx >= tot) return;
  int i = (int)(idx / H);
  int d = (int)(idx % H);
  float s = dinv[i];
  out[idx] = hw[idx] * s * s + bias[d];
}
__global__ void k_scatter(const float* __restrict__ hw, const int* __restrict__ ei,
                          const float* __restrict__ dinv, float* __restrict__ out,
                          int E, int H) {
  long long idx = (long long)blockIdx.x * blockDim.x + threadIdx.x;
  int per = H >> 2;
  long long tot = (long long)E * per;
  if (idx >= tot) return;
  int e  = (int)(idx / per);
  int dq = (int)(idx % per) << 2;
  int rs = ei[e];
  int cd = ei[E + e];
  float nm = dinv[rs] * dinv[cd];
  float4 v = *(const float4*)(hw + (size_t)rs * H + dq);
  float* o = out + (size_t)cd * H + dq;
  atomicAdd(o + 0, v.x * nm);
  atomicAdd(o + 1, v.y * nm);
  atomicAdd(o + 2, v.z * nm);
  atomicAdd(o + 3, v.w * nm);
}
__global__ void k_zero(float* p, int n) {
  int i = blockIdx.x * blockDim.x + threadIdx.x;
  if (i < n) p[i] = 0.0f;
}
__global__ void k_cnt(const int* __restrict__ batch, float* cnts, int n) {
  int i = blockIdx.x * blockDim.x + threadIdx.x;
  if (i < n) atomicAdd(&cnts[batch[i]], 1.0f);
}
__global__ void k_pool(const float* __restrict__ h, const int* __restrict__ batch,
                       float* __restrict__ sums, int n, int H) {
  long long idx = (long long)blockIdx.x * blockDim.x + threadIdx.x;
  int per = H >> 2;
  long long tot = (long long)n * per;
  if (idx >= tot) return;
  int i  = (int)(idx / per);
  int dq = (int)(idx % per) << 2;
  int g  = batch[i];
  float4 v = *(const float4*)(h + (size_t)i * H + dq);
  float* s = sums + (size_t)g * H + dq;
  atomicAdd(s + 0, v.x);
  atomicAdd(s + 1, v.y);
  atomicAdd(s + 2, v.z);
  atomicAdd(s + 3, v.w);
}
__global__ void k_div(const float* __restrict__ sums, const float* __restrict__ cnts,
                      float* __restrict__ hg, int G, int H) {
  int idx = blockIdx.x * blockDim.x + threadIdx.x;
  if (idx >= G * H) return;
  hg[idx] = sums[idx] / fmaxf(cnts[idx / H], 1.0f);
}

// ---------------- host orchestration ----------------
static inline unsigned cdivll(long long a, long long b) { return (unsigned)((a + b - 1) / b); }
static inline size_t al16(size_t x) { return (x + 15) & ~(size_t)15; }

extern "C" void kernel_launch(void* const* d_in, const int* in_sizes, int n_in,
                              void* d_out, int out_size, void* d_ws, size_t ws_size,
                              hipStream_t stream) {
  const float* x     = (const float*)d_in[0];
  const int*   ei    = (const int*)d_in[1];
  const int*   batch = (const int*)d_in[2];
  const float* W1  = (const float*)d_in[3];
  const float* b1  = (const float*)d_in[4];
  const float* W2  = (const float*)d_in[5];
  const float* b2  = (const float*)d_in[6];
  const float* Wg1 = (const float*)d_in[7];
  const float* bg1 = (const float*)d_in[8];
  const float* Wg2 = (const float*)d_in[9];
  const float* bg2 = (const float*)d_in[10];
  const float* Wc  = (const float*)d_in[11];
  const float* bc  = (const float*)d_in[12];

  const int IN = 400, H = 768, OUT = 78, G = 1024;
  const int n = in_sizes[0] / IN;             // 100000
  const int E = in_sizes[1] / 2;              // 100000
  const int Mp  = ((n + 127) / 128) * 128;    // 100096
  const int INp = ((IN + 31) / 32) * 32;      // 416
  const int OUTp = 128;                       // padded classifier N

  // workspace layout (bytes, 16B aligned). xb overlays B0f (dead before G3).
  char* base = (char*)d_ws;
  size_t off = 0;
  size_t szB0  = al16((size_t)n * H * 4);
  size_t szXb  = al16((size_t)Mp * INp * 2);
  float*          B0f = (float*)(base + off);
  unsigned short* xb  = (unsigned short*)(base + off);
  off += (szB0 > szXb ? szB0 : szXb);
  float*          B1f = (float*)(base + off);          off += al16((size_t)n * H * 4);
  unsigned short* h1b = (unsigned short*)(base + off); off += al16((size_t)Mp * H * 2); // reused as h2b
  unsigned short* hb  = (unsigned short*)(base + off); off += al16((size_t)Mp * H * 2);
  unsigned short* W1t = (unsigned short*)(base + off); off += al16((size_t)H * INp * 2);
  unsigned short* W2t = (unsigned short*)(base + off); off += al16((size_t)H * H * 2);
  unsigned short* Wg1t= (unsigned short*)(base + off); off += al16((size_t)H * H * 2);
  unsigned short* Wg2t= (unsigned short*)(base + off); off += al16((size_t)H * H * 2);
  unsigned short* WcT = (unsigned short*)(base + off); off += al16((size_t)OUTp * H * 2);
  float* deg  = (float*)(base + off); off += al16((size_t)n * 4);
  float* dinv = (float*)(base + off); off += al16((size_t)n * 4);
  float* sums = (float*)(base + off); off += al16((size_t)G * H * 4);
  float* cnts = (float*)(base + off); off += al16((size_t)G * 4);
  float* hg   = (float*)(base + off); off += al16((size_t)G * H * 4);
  unsigned short* hgb = (unsigned short*)(base + off); off += al16((size_t)G * H * 2);

  float* out_logits = (float*)d_out;
  float* out_h      = out_logits + (size_t)G * OUT;

  dim3 blk(256);
  unsigned short* h2b = h1b;

  // ---- pack weights + input to bf16 (transposed [N][K], zero padded) ----
  k_pack_rows     <<<cdivll((long long)Mp * INp, 256), blk, 0, stream>>>(x, xb, n, IN, Mp, INp);
  k_pack_transpose<<<cdivll((long long)H * INp, 256), blk, 0, stream>>>(W1, W1t, IN, H, INp, H);
  k_pack_transpose<<<cdivll((long long)H * H, 256), blk, 0, stream>>>(W2,  W2t,  H, H, H, H);
  k_pack_transpose<<<cdivll((long long)H * H, 256), blk, 0, stream>>>(Wg1, Wg1t, H, H, H, H);
  k_pack_transpose<<<cdivll((long long)H * H, 256), blk, 0, stream>>>(Wg2, Wg2t, H, H, H, H);
  k_pack_transpose<<<cdivll((long long)OUTp * H, 256), blk, 0, stream>>>(Wc, WcT, H, OUT, H, OUTp);

  // degrees (shared by both conv layers)
  k_init_deg <<<cdivll(n, 256), blk, 0, stream>>>(deg, n);
  k_count_deg<<<cdivll(E, 256), blk, 0, stream>>>(ei, deg, E);
  k_rsqrt    <<<cdivll(n, 256), blk, 0, stream>>>(deg, dinv, n);

  // 1) h1 = relu(x@W1 + b1) -> h1b (bf16)
  { dim3 g(H / BN, Mp / BM);
    gemm_bf16_wmma<<<g, blk, 0, stream>>>(xb, W1t, b1, h1b, Mp, H, INp, Mp, H, 1, 1, 1); }
  // 2) h = h1@W2 + b2 -> hb (bf16)
  { dim3 g(H / BN, Mp / BM);
    gemm_bf16_wmma<<<g, blk, 0, stream>>>(h1b, W2t, b2, hb, Mp, H, H, Mp, H, 0, 1, 1); }
  // 3) conv1: hw = h@Wg1 -> B0f (f32, clobbers xb) ; aggregate -> B1f
  { dim3 g(H / BN, Mp / BM);
    gemm_bf16_wmma<<<g, blk, 0, stream>>>(hb, Wg1t, nullptr, B0f, n, H, H, Mp, H, 0, 0, 0); }
  k_self_bias<<<cdivll((long long)n * H, 256), blk, 0, stream>>>(B0f, dinv, bg1, B1f, n, H);
  k_scatter  <<<cdivll((long long)E * (H / 4), 256), blk, 0, stream>>>(B0f, ei, dinv, B1f, E, H);
  // pack h2 -> bf16 (h2b reuses h1b storage)
  k_pack_rows<<<cdivll((long long)Mp * H, 256), blk, 0, stream>>>(B1f, h2b, n, H, Mp, H);
  // 4) conv2: hw = h2@Wg2 -> B0f ; aggregate -> out_h (second output)
  { dim3 g(H / BN, Mp / BM);
    gemm_bf16_wmma<<<g, blk, 0, stream>>>(h2b, Wg2t, nullptr, B0f, n, H, H, Mp, H, 0, 0, 0); }
  k_self_bias<<<cdivll((long long)n * H, 256), blk, 0, stream>>>(B0f, dinv, bg2, out_h, n, H);
  k_scatter  <<<cdivll((long long)E * (H / 4), 256), blk, 0, stream>>>(B0f, ei, dinv, out_h, E, H);

  // 5) global mean pool
  k_zero<<<cdivll(G * H + G, 256), blk, 0, stream>>>(sums, G * H + G); // sums + cnts contiguous
  k_cnt <<<cdivll(n, 256), blk, 0, stream>>>(batch, cnts, n);
  k_pool<<<cdivll((long long)n * (H / 4), 256), blk, 0, stream>>>(out_h, batch, sums, n, H);
  k_div <<<cdivll(G * H, 256), blk, 0, stream>>>(sums, cnts, hg, G, H);
  k_pack_rows<<<cdivll((long long)G * H, 256), blk, 0, stream>>>(hg, hgb, G, H, G, H);

  // 6) logits = hg@Wc + bc -> first output (N=78 handled by store guard)
  { dim3 g(1, G / BM);
    gemm_bf16_wmma<<<g, blk, 0, stream>>>(hgb, WcT, bc, out_logits, G, OUT, H, G, OUTp, 0, 1, 0); }
}
